// SPVCNNBackbone_71116068488050
// MI455X (gfx1250) — matrix operand order repaired and checked by reading, not compiled, not run on hardware
//
#include <hip/hip_runtime.h>
#include <hip/hip_bf16.h>

// ---------------------------------------------------------------------------
// SPVCNN backbone for MI455X (gfx1250, wave32, WMMA).
// Heavy GEMMs (implicit-GEMM 3D convs + refine linears) run on
// v_wmma_f32_16x16x32_f16 with f32 accumulation, with 4-wide tile reuse so
// each loaded fragment feeds 4 WMMAs. Biases dropped (cancel under BN).
// ---------------------------------------------------------------------------

typedef __attribute__((ext_vector_type(16))) _Float16 v16h;
typedef __attribute__((ext_vector_type(8)))  float    v8f;

union HFrag {
    v16h   v;
    float4 f4[2];
};

__device__ __forceinline__ float4 f4zero() {
    float4 z; z.x = 0.f; z.y = 0.f; z.z = 0.f; z.w = 0.f; return z;
}

__device__ __forceinline__ v8f wmma_f16(v16h a, v16h b, v8f c) {
    return __builtin_amdgcn_wmma_f32_16x16x32_f16(
        /*neg_a=*/false, a, /*neg_b=*/false, b,
        /*c_mod=*/(short)0, c, /*reuse_a=*/false, /*reuse_b=*/false);
}

// ---------------------------------------------------------------------------
// Small utility kernels
// ---------------------------------------------------------------------------
__global__ void fill_u32_kernel(unsigned int* __restrict__ p, unsigned int val, int n) {
    int t = blockIdx.x * blockDim.x + threadIdx.x;
    if (t < n) p[t] = val;
}

__global__ void copy_f32_kernel(const float* __restrict__ s, float* __restrict__ d, int n) {
    int t = blockIdx.x * blockDim.x + threadIdx.x;
    if (t < n) d[t] = s[t];
}

// W (K x N) f32 row-major  ->  Wt (N x K) f16 (GEMM-B transposed, contiguous K)
__global__ void prep_lin_kernel(const float* __restrict__ W, _Float16* __restrict__ Wt,
                                int K, int N) {
    int t = blockIdx.x * blockDim.x + threadIdx.x;
    if (t >= N * K) return;
    int n = t / K, k = t % K;
    Wt[t] = (_Float16)W[k * N + n];
}

// W (co, ci, taps) f32 -> Wt (taps, co, ci) f16 (ci contiguous)
__global__ void prep_conv_kernel(const float* __restrict__ W, _Float16* __restrict__ Wt,
                                 int ci, int co, int taps) {
    int t = blockIdx.x * blockDim.x + threadIdx.x;
    if (t >= taps * co * ci) return;
    int tap = t / (co * ci);
    int rem = t % (co * ci);
    int o = rem / ci, i = rem % ci;
    Wt[t] = (_Float16)W[(o * ci + i) * taps + tap];
}

// ---------------------------------------------------------------------------
// Point MLP: y = xyz @ W (3x32), no bias (cancels under BN)
// ---------------------------------------------------------------------------
__global__ void point_mlp_kernel(const float* __restrict__ xyz, const float* __restrict__ W,
                                 float* __restrict__ out, int n) {
    int r = blockIdx.x * blockDim.x + threadIdx.x;
    if (r >= n) return;
    float x0 = xyz[r * 3 + 0], x1 = xyz[r * 3 + 1], x2 = xyz[r * 3 + 2];
    float* orow = out + (size_t)r * 32;
    #pragma unroll
    for (int c = 0; c < 32; ++c)
        orow[c] = x0 * W[c] + x1 * W[32 + c] + x2 * W[64 + c];
}

// ---------------------------------------------------------------------------
// BN batch statistics: per-channel sum & sum-of-squares (src is M x C)
// stats[0..C) = sum, stats[C..2C) = sumsq
// ---------------------------------------------------------------------------
__global__ void stats_kernel(const float* __restrict__ src, float* __restrict__ stats,
                             int M, int C) {
    int t = blockIdx.x * blockDim.x + threadIdx.x;
    if (t >= C * 256) return;
    int c  = t % C;
    int r0 = t / C;
    float s = 0.f, ss = 0.f;
    for (int r = r0; r < M; r += 256) {
        float v = src[(size_t)r * C + c];
        s += v; ss += v * v;
    }
    atomicAdd(&stats[c], s);
    atomicAdd(&stats[C + c], ss);
}

// Apply BN (+optional ReLU); optionally write f16 and/or f32 copies.
__global__ void bn_apply_kernel(const float* __restrict__ src, const float* __restrict__ stats,
                                const float* __restrict__ g, const float* __restrict__ b,
                                _Float16* __restrict__ dst16, float* __restrict__ dst32,
                                int M, int C, int relu) {
    int t = blockIdx.x * blockDim.x + threadIdx.x;
    if (t >= M * C) return;
    int c = t % C;
    float invM = 1.0f / (float)M;
    float mean = stats[c] * invM;
    float var  = stats[C + c] * invM - mean * mean;
    float y = (src[t] - mean) * rsqrtf(var + 1e-5f) * g[c] + b[c];
    if (relu) y = fmaxf(y, 0.f);
    if (dst16) dst16[t] = (_Float16)y;
    if (dst32) dst32[t] = y;
}

// out = relu(bn2(y2) + bnsc(ysc)) -> f16 voxel features for next stage
__global__ void combine_kernel(const float* __restrict__ y2, const float* __restrict__ st2,
                               const float* __restrict__ g2, const float* __restrict__ b2,
                               const float* __restrict__ ys, const float* __restrict__ sts,
                               const float* __restrict__ gs, const float* __restrict__ bs,
                               _Float16* __restrict__ dst, int M, int C) {
    int t = blockIdx.x * blockDim.x + threadIdx.x;
    if (t >= M * C) return;
    int c = t % C;
    float invM = 1.0f / (float)M;
    float m2 = st2[c] * invM, v2 = st2[C + c] * invM - m2 * m2;
    float ms = sts[c] * invM, vs = sts[C + c] * invM - ms * ms;
    float a = (y2[t] - m2) * rsqrtf(v2 + 1e-5f) * g2[c] + b2[c];
    float s = (ys[t] - ms) * rsqrtf(vs + 1e-5f) * gs[c] + bs[c];
    dst[t] = (_Float16)fmaxf(a + s, 0.f);
}

// ---------------------------------------------------------------------------
// Voxelization
// ---------------------------------------------------------------------------
__global__ void voxel_min_kernel(const float* __restrict__ xyz, int* __restrict__ mins, int n) {
    int r = blockIdx.x * blockDim.x + threadIdx.x;
    if (r >= n) return;
    #pragma unroll
    for (int d = 0; d < 3; ++d) {
        int v = (int)floorf(xyz[r * 3 + d] / 0.05f);
        atomicMin(&mins[d], v);
    }
}

__global__ void voxel_assign_kernel(const float* __restrict__ xyz, const int* __restrict__ mins,
                                    int* __restrict__ scat, int n) {
    int r = blockIdx.x * blockDim.x + threadIdx.x;
    if (r >= n) return;
    int vc[3]; bool valid = true;
    #pragma unroll
    for (int d = 0; d < 3; ++d) {
        vc[d] = (int)floorf(xyz[r * 3 + d] / 0.05f) - mins[d];
        valid = valid && (vc[d] < 20);
    }
    int local = vc[0] * 400 + vc[1] * 20 + vc[2];
    if (local < 0) local = 0;
    if (local > 7999) local = 7999;
    scat[r] = valid ? local : -1;
}

__global__ void scatter_kernel(const float* __restrict__ pf, const int* __restrict__ scat,
                               float* __restrict__ vsum, float* __restrict__ vcnt, int n) {
    int t = blockIdx.x * blockDim.x + threadIdx.x;
    if (t >= n * 32) return;
    int r = t >> 5, c = t & 31;
    int s = scat[r];
    if (s >= 0) {
        atomicAdd(&vsum[s * 32 + c], pf[t]);
        if (c == 0) atomicAdd(&vcnt[s], 1.f);
    }
}

__global__ void vox_final_kernel(const float* __restrict__ vsum, const float* __restrict__ vcnt,
                                 _Float16* __restrict__ dst) {
    int t = blockIdx.x * blockDim.x + threadIdx.x;
    if (t >= 8000 * 32) return;
    float cnt = fmaxf(vcnt[t >> 5], 1.f);
    dst[t] = (_Float16)(vsum[t] / cnt);
}

// ---------------------------------------------------------------------------
// Implicit-GEMM 3D conv on the 20^3 grid (SAME padding), WMMA f16->f32.
// Each wave computes a 16x64 output tile: the A (activation) fragment is
// loaded once per (tap, k-step) and reused across 4 weight fragments.
// A: (8000 x ci) f16, Wt: (taps, co, ci) f16, out: (8000 x co) f32.
// grid = (500, co/64), block = 32.
// ---------------------------------------------------------------------------
__global__ __launch_bounds__(32)
void conv_wmma_kernel(const _Float16* __restrict__ A, const _Float16* __restrict__ Wt,
                      float* __restrict__ out, int ci, int co, int taps) {
    const int lane = threadIdx.x;
    const int lmod = lane & 15;
    const int hs   = lane >> 4;
    const int kbA  = hs * 8;    // A fragment: K runs at kbA and kbA+16
    const int kbB  = hs * 8;    // B fragment: K runs at kbB and kbB+16
    const int row  = blockIdx.x * 16 + lmod;
    const int vx = row / 400, vy = (row / 20) % 20, vz = row % 20;
    const int col0 = blockIdx.y * 64 + lmod;   // + j*16 per accumulator

    v8f acc0 = {}, acc1 = {}, acc2 = {}, acc3 = {};
    for (int t = 0; t < taps; ++t) {
        int dx = 0, dy = 0, dz = 0;
        if (taps == 27) { dx = t / 9 - 1; dy = (t / 3) % 3 - 1; dz = t % 3 - 1; }
        const int nx = vx + dx, ny = vy + dy, nz = vz + dz;
        const bool ok = ((unsigned)nx < 20u) && ((unsigned)ny < 20u) && ((unsigned)nz < 20u);
        const int nrow = ok ? (nx * 400 + ny * 20 + nz) : 0;
        const _Float16* arow = A + (size_t)nrow * ci;
        const _Float16* brow = Wt + ((size_t)t * co + (size_t)col0) * ci;
        const size_t bstep = (size_t)16 * ci;   // 16 output channels per sub-tile

        for (int k0 = 0; k0 < ci; k0 += 32) {
            HFrag a;
            if (ok) {
                a.f4[0] = *(const float4*)(arow + k0 + kbA);
                a.f4[1] = *(const float4*)(arow + k0 + kbA + 16);
            } else {
                a.f4[0] = f4zero();
                a.f4[1] = f4zero();
            }
            HFrag b0, b1, b2, b3;
            b0.f4[0] = *(const float4*)(brow + 0 * bstep + k0 + kbB);
            b0.f4[1] = *(const float4*)(brow + 0 * bstep + k0 + kbB + 16);
            b1.f4[0] = *(const float4*)(brow + 1 * bstep + k0 + kbB);
            b1.f4[1] = *(const float4*)(brow + 1 * bstep + k0 + kbB + 16);
            b2.f4[0] = *(const float4*)(brow + 2 * bstep + k0 + kbB);
            b2.f4[1] = *(const float4*)(brow + 2 * bstep + k0 + kbB + 16);
            b3.f4[0] = *(const float4*)(brow + 3 * bstep + k0 + kbB);
            b3.f4[1] = *(const float4*)(brow + 3 * bstep + k0 + kbB + 16);
            acc0 = wmma_f16(a.v, b0.v, acc0);
            acc1 = wmma_f16(a.v, b1.v, acc1);
            acc2 = wmma_f16(a.v, b2.v, acc2);
            acc3 = wmma_f16(a.v, b3.v, acc3);
        }
    }

    float* orow = out + (size_t)(blockIdx.x * 16 + hs * 8) * co + col0;
    #pragma unroll
    for (int r = 0; r < 8; ++r) {
        orow[(size_t)r * co +  0] = acc0[r];
        orow[(size_t)r * co + 16] = acc1[r];
        orow[(size_t)r * co + 32] = acc2[r];
        orow[(size_t)r * co + 48] = acc3[r];
    }
}

// ---------------------------------------------------------------------------
// Refine GEMM with fused voxel->point gather + concat.
// Each wave computes a 64x16 output tile: the B (weight) fragment is loaded
// once per k-step and reused across 4 row sub-tiles.
// A row k<c comes from vf[scat[row]] (0 if invalid), k>=c from feats[row].
// Wt: (co x (c+cp)) f16. out: (n x co) f32.
// grid = (ceil(n/64), co/16), block = 32.
// ---------------------------------------------------------------------------
__global__ __launch_bounds__(32)
void refine_wmma_kernel(const _Float16* __restrict__ vf, const _Float16* __restrict__ feats,
                        const int* __restrict__ scat, const _Float16* __restrict__ Wt,
                        float* __restrict__ out, int n, int c, int cp, int co) {
    const int K = c + cp;
    const int lane = threadIdx.x;
    const int lmod = lane & 15;
    const int hs   = lane >> 4;
    const int kbA  = hs * 8;
    const int kbB  = hs * 8;
    const _Float16* brow = Wt + (size_t)(blockIdx.y * 16 + lmod) * K;

    // Per-row gather state for the 4 row sub-tiles handled by this wave.
    int  rows[4];
    bool inb[4];
    int  sc[4];
    const _Float16* prow[4];
    const _Float16* frow[4];
    #pragma unroll
    for (int m = 0; m < 4; ++m) {
        rows[m] = blockIdx.x * 64 + m * 16 + lmod;
        inb[m]  = rows[m] < n;
        sc[m]   = inb[m] ? scat[rows[m]] : -1;
        prow[m] = vf + (size_t)(sc[m] < 0 ? 0 : sc[m]) * c;
        frow[m] = feats + (size_t)(inb[m] ? rows[m] : 0) * cp;
    }

    v8f acc[4] = {{}, {}, {}, {}};
    for (int k0 = 0; k0 < K; k0 += 32) {
        HFrag b;
        b.f4[0] = *(const float4*)(brow + k0 + kbB);
        b.f4[1] = *(const float4*)(brow + k0 + kbB + 16);
        const int kg0 = k0 + kbA;
        const int kg1 = k0 + kbA + 16;
        #pragma unroll
        for (int m = 0; m < 4; ++m) {
            HFrag a;
            if (kg0 < c) a.f4[0] = (sc[m] >= 0) ? *(const float4*)(prow[m] + kg0) : f4zero();
            else         a.f4[0] = inb[m] ? *(const float4*)(frow[m] + (kg0 - c)) : f4zero();
            if (kg1 < c) a.f4[1] = (sc[m] >= 0) ? *(const float4*)(prow[m] + kg1) : f4zero();
            else         a.f4[1] = inb[m] ? *(const float4*)(frow[m] + (kg1 - c)) : f4zero();
            acc[m] = wmma_f16(a.v, b.v, acc[m]);
        }
    }

    #pragma unroll
    for (int m = 0; m < 4; ++m) {
        #pragma unroll
        for (int r = 0; r < 8; ++r) {
            int orow = blockIdx.x * 64 + m * 16 + hs * 8 + r;
            if (orow < n)
                out[(size_t)orow * co + blockIdx.y * 16 + lmod] = acc[m][r];
        }
    }
}

// ---------------------------------------------------------------------------
// Host side
// ---------------------------------------------------------------------------
static inline int g256(int n) { return (n + 255) / 256; }

extern "C" void kernel_launch(void* const* d_in, const int* in_sizes, int n_in,
                              void* d_out, int out_size, void* d_ws, size_t ws_size,
                              hipStream_t stream) {
    (void)in_sizes; (void)n_in; (void)out_size; (void)ws_size;
    const int N = 100000;

    // Input index map (jax pytree: dicts flatten by sorted keys, lists in order).
    // blocks[i] base = i*12:
    //   +0 bn1.b +1 bn1.g +2 bn2.b +3 bn2.g +4 conv1.W +5 conv1.b
    //   +6 conv2.W +7 conv2.b +8 sc_bn.b +9 sc_bn.g +10 sc_conv.W +11 sc_conv.b
    // 36..39: point_mlp bn.b, bn.g, lin.W, lin.b
    // 40+4i:  refine[i] bn.b, bn.g, lin.W, lin.b
    // 52: xyz
    const float* xyz = (const float*)d_in[52];

    // --- workspace arena (deterministic bump allocator) ---
    char* wsbase = (char*)d_ws;
    size_t off = 0;
    auto alloc = [&](size_t bytes) -> char* {
        char* p = wsbase + off;
        off += (bytes + 255) & ~(size_t)255;
        return p;
    };
    int*       mins   = (int*)alloc(16);
    float*     statsA = (float*)alloc(2 * 256 * 4);
    float*     statsB = (float*)alloc(2 * 256 * 4);
    float*     vsum   = (float*)alloc((size_t)8000 * 32 * 4);
    float*     vcnt   = (float*)alloc((size_t)8000 * 4);
    _Float16*  vbA    = (_Float16*)alloc((size_t)8000 * 256 * 2);
    _Float16*  vbB    = (_Float16*)alloc((size_t)8000 * 256 * 2);
    _Float16*  vbH    = (_Float16*)alloc((size_t)8000 * 256 * 2);
    float*     yv1    = (float*)alloc((size_t)8000 * 256 * 4);
    float*     yv2    = (float*)alloc((size_t)8000 * 256 * 4);
    int*       scat   = (int*)alloc((size_t)N * 4);
    float*     pf32   = (float*)alloc((size_t)N * 32 * 4);
    _Float16*  fbA    = (_Float16*)alloc((size_t)N * 256 * 2);
    _Float16*  fbB    = (_Float16*)alloc((size_t)N * 256 * 2);
    float*     yp     = (float*)alloc((size_t)N * 256 * 4);
    _Float16*  wt1    = (_Float16*)alloc((size_t)27 * 256 * 256 * 2);
    _Float16*  wt2    = (_Float16*)alloc((size_t)27 * 256 * 256 * 2);
    _Float16*  wtsc   = (_Float16*)alloc((size_t)256 * 256 * 2);
    _Float16*  wtr    = (_Float16*)alloc((size_t)384 * 256 * 2);

    // --- output part 1: xyz passthrough ---
    copy_f32_kernel<<<g256(300000), 256, 0, stream>>>(xyz, (float*)d_out, 300000);

    // --- point MLP + BN(+relu) ---
    point_mlp_kernel<<<g256(N), 256, 0, stream>>>(xyz, (const float*)d_in[38], yp, N);
    fill_u32_kernel<<<1, 256, 0, stream>>>((unsigned*)statsA, 0u, 2 * 32);
    stats_kernel<<<g256(32 * 256), 256, 0, stream>>>(yp, statsA, N, 32);
    bn_apply_kernel<<<g256(N * 32), 256, 0, stream>>>(
        yp, statsA, (const float*)d_in[37], (const float*)d_in[36],
        fbA, pf32, N, 32, 1);

    // --- voxelization + scatter-mean ---
    fill_u32_kernel<<<1, 32, 0, stream>>>((unsigned*)mins, 0x7fffffffu, 3);
    voxel_min_kernel<<<g256(N), 256, 0, stream>>>(xyz, mins, N);
    voxel_assign_kernel<<<g256(N), 256, 0, stream>>>(xyz, mins, scat, N);
    fill_u32_kernel<<<g256(8000 * 32), 256, 0, stream>>>((unsigned*)vsum, 0u, 8000 * 32);
    fill_u32_kernel<<<g256(8000), 256, 0, stream>>>((unsigned*)vcnt, 0u, 8000);
    scatter_kernel<<<g256(N * 32), 256, 0, stream>>>(pf32, scat, vsum, vcnt, N);
    vox_final_kernel<<<g256(8000 * 32), 256, 0, stream>>>(vsum, vcnt, vbA);

    // --- residual blocks + refine layers ---
    const int CI[3] = {32, 64, 128};
    const int CO[3] = {64, 128, 256};
    _Float16* vX = vbA; _Float16* vY = vbB;
    _Float16* fX = fbA; _Float16* fY = fbB;

    for (int i = 0; i < 3; ++i) {
        const int ci = CI[i], co = CO[i];
        const int bb = i * 12;

        prep_conv_kernel<<<g256(27 * co * ci), 256, 0, stream>>>((const float*)d_in[bb + 4],  wt1, ci, co, 27);
        prep_conv_kernel<<<g256(27 * co * co), 256, 0, stream>>>((const float*)d_in[bb + 6],  wt2, co, co, 27);
        prep_conv_kernel<<<g256(co * ci),      256, 0, stream>>>((const float*)d_in[bb + 10], wtsc, ci, co, 1);

        // h = relu(bn1(conv1(x)))
        conv_wmma_kernel<<<dim3(500, co / 64), 32, 0, stream>>>(vX, wt1, yv1, ci, co, 27);
        fill_u32_kernel<<<1, 256, 0, stream>>>((unsigned*)statsA, 0u, 2 * co);
        stats_kernel<<<g256(co * 256), 256, 0, stream>>>(yv1, statsA, 8000, co);
        bn_apply_kernel<<<g256(8000 * co), 256, 0, stream>>>(
            yv1, statsA, (const float*)d_in[bb + 1], (const float*)d_in[bb + 0],
            vbH, nullptr, 8000, co, 1);

        // y2 = conv2(h); ysc = sc_conv(x)
        conv_wmma_kernel<<<dim3(500, co / 64), 32, 0, stream>>>(vbH, wt2, yv2, co, co, 27);
        conv_wmma_kernel<<<dim3(500, co / 64), 32, 0, stream>>>(vX, wtsc, yv1, ci, co, 1);
        fill_u32_kernel<<<1, 256, 0, stream>>>((unsigned*)statsA, 0u, 2 * co);
        fill_u32_kernel<<<1, 256, 0, stream>>>((unsigned*)statsB, 0u, 2 * co);
        stats_kernel<<<g256(co * 256), 256, 0, stream>>>(yv2, statsA, 8000, co);
        stats_kernel<<<g256(co * 256), 256, 0, stream>>>(yv1, statsB, 8000, co);

        // vf_next = relu(bn2(y2) + sc_bn(ysc))
        combine_kernel<<<g256(8000 * co), 256, 0, stream>>>(
            yv2, statsA, (const float*)d_in[bb + 3], (const float*)d_in[bb + 2],
            yv1, statsB, (const float*)d_in[bb + 9], (const float*)d_in[bb + 8],
            vY, 8000, co);

        // refine: feats = relu(bn(cat([gather(vf_next), feats]) @ W))
        const int rb = 40 + 4 * i;
        const int K = ci + co;  // cat = [pt (co), feats (ci)]
        prep_lin_kernel<<<g256(co * K), 256, 0, stream>>>((const float*)d_in[rb + 2], wtr, K, co);
        refine_wmma_kernel<<<dim3((N + 63) / 64, co / 16), 32, 0, stream>>>(
            vY, fX, scat, wtr, yp, N, co, ci, co);
        fill_u32_kernel<<<1, 256, 0, stream>>>((unsigned*)statsA, 0u, 2 * co);
        stats_kernel<<<g256(co * 256), 256, 0, stream>>>(yp, statsA, N, co);
        const bool last = (i == 2);
        bn_apply_kernel<<<g256(N * co), 256, 0, stream>>>(
            yp, statsA, (const float*)d_in[rb + 1], (const float*)d_in[rb + 0],
            last ? (_Float16*)nullptr : fY,
            last ? ((float*)d_out + 300000) : (float*)nullptr,
            N, co, 1);

        _Float16* tv = vX; vX = vY; vY = tv;
        _Float16* tf = fX; fX = fY; fY = tf;
    }
}
